// Head_9972914061388
// MI455X (gfx1250) — matrix operand order repaired and checked by reading, not compiled, not run on hardware
//
#include <hip/hip_runtime.h>
#include <hip/hip_bf16.h>

// ---------------------------------------------------------------------------
// Causal single-head attention for MI455X (gfx1250, wave32, WMMA).
//   B=8, T=2048, C=1024, H=128
// Pipeline:
//   k_wtrans : Wq/Wk/Wv (f32, [C,H]) -> Wt (bf16, [3][H][C])  (transposed)
//   k_proj   : x @ W{q,k,v} via v_wmma_f32_16x16x32_bf16
//              -> qb, kb (bf16 row-major [B*T,H]) and vt (bf16 [B][H][T], V^T)
//   k_attn   : per-wave flash attention, 16 query rows / wave, online softmax
//
// Fragment layouts (CDNA5 wave32, v_wmma_f32_16x16x32_bf16):
//   A (16x32, MxK): lane = M%16; elements 0..7  = K{0..7} (lanes 0-15)
//                                             or K{8..15} (lanes 16-31);
//                   elements 8..15 = K{16..23} / K{24..31}.   (interleaved)
//   B (32x16, KxN): lane = N%16; elements 0..7  = K{0..7}  (lanes 0-15)
//                                             or K{8..15} (lanes 16-31); 8..15 = K{16..23}/K{24..31}.
//   C/D (16x16 f32): VGPR r, lane L -> (row r + 8*(L/16), col L%16).
// ---------------------------------------------------------------------------

typedef __bf16 bf16;
typedef __attribute__((ext_vector_type(8)))  bf16  v8bf;
typedef __attribute__((ext_vector_type(16))) bf16  v16bf;
typedef __attribute__((ext_vector_type(8)))  float v8f;
typedef __attribute__((ext_vector_type(4)))  float v4f;

#define NB   8
#define NT   2048
#define NC   1024
#define NH   128
#define WT_ELEMS   (3 * NH * NC)        // 393216
#define QK_ELEMS   (NB * NT * NH)       // 2097152

__device__ __forceinline__ v8f wmma_bf16(v16bf a, v16bf b, v8f c) {
    return __builtin_amdgcn_wmma_f32_16x16x32_bf16(
        /*neg_a=*/false, a, /*neg_b=*/false, b,
        /*c_mod=*/(short)0, c, /*reuse_a=*/false, /*reuse_b=*/false);
}

// A-operand fragment: interleaved K split. Caller passes
// p0 = base + (hi?8:0), p1 = base + 16 + (hi?8:0).
__device__ __forceinline__ v16bf ldfragA(const bf16* p0, const bf16* p1) {
    v8bf a = *(const v8bf*)p0;
    v8bf b = *(const v8bf*)p1;
    v16bf r;
#pragma unroll
    for (int i = 0; i < 8; ++i) { r[i] = a[i]; r[i + 8] = b[i]; }
    return r;
}

// B-operand fragment: interleaved K split (matches A).
// Lane group lo -> K 0..7,16..23; group hi -> K 8..15,24..31.
__device__ __forceinline__ v16bf ldfragB(const bf16* p, int hi) {
    const v8bf* q = (const v8bf*)(p + hi * 8);
    v8bf a = q[0];
    v8bf b = q[2];
    v16bf r;
#pragma unroll
    for (int i = 0; i < 8; ++i) { r[i] = a[i]; r[i + 8] = b[i]; }
    return r;
}

// A-operand fragment from f32 source (convert on the fly).
__device__ __forceinline__ v16bf ldfragA_f32(const float* p0, const float* p1) {
    v4f a0 = *(const v4f*)p0;
    v4f a1 = *(const v4f*)(p0 + 4);
    v4f b0 = *(const v4f*)p1;
    v4f b1 = *(const v4f*)(p1 + 4);
    v16bf r;
#pragma unroll
    for (int i = 0; i < 4; ++i) {
        r[i]      = (bf16)a0[i];
        r[4 + i]  = (bf16)a1[i];
        r[8 + i]  = (bf16)b0[i];
        r[12 + i] = (bf16)b1[i];
    }
    return r;
}

__device__ __forceinline__ float rmax16(float v) {
#pragma unroll
    for (int m = 1; m < 16; m <<= 1) v = fmaxf(v, __shfl_xor(v, m, 32));
    return v;
}
__device__ __forceinline__ float rsum16(float v) {
#pragma unroll
    for (int m = 1; m < 16; m <<= 1) v += __shfl_xor(v, m, 32);
    return v;
}

// ---------------------------------------------------------------------------
// Kernel 1: transpose weights to bf16  Wt[m][h][c] = W_m[c][h]
// ---------------------------------------------------------------------------
__global__ __launch_bounds__(256) void k_wtrans(const float* __restrict__ Wq,
                                                const float* __restrict__ Wk,
                                                const float* __restrict__ Wv,
                                                bf16* __restrict__ wt) {
    int gid = blockIdx.x * 256 + threadIdx.x;
    if (gid >= WT_ELEMS) return;
    int m   = gid >> 17;          // / (128*1024)
    int rem = gid & 131071;
    int h   = rem >> 10;
    int c   = rem & 1023;
    const float* W = (m == 0) ? Wq : ((m == 1) ? Wk : Wv);
    wt[gid] = (bf16)W[c * NH + h];
}

// ---------------------------------------------------------------------------
// Kernel 2: QKV projection. One wave = (16-row x tile, one of q/k/v).
// 3072 waves total -> 384 blocks of 256 threads (8 waves).
// ---------------------------------------------------------------------------
__global__ __launch_bounds__(256) void k_proj(const float* __restrict__ x,
                                              const bf16* __restrict__ wt,
                                              bf16* __restrict__ qb,
                                              bf16* __restrict__ kb,
                                              bf16* __restrict__ vt) {
    int wave   = (blockIdx.x * 256 + threadIdx.x) >> 5;
    int lane   = threadIdx.x & 31;
    int lanelo = lane & 15;
    int hi     = lane >> 4;

    int m    = wave >> 10;        // 0=q, 1=k, 2=v
    int rt   = wave & 1023;       // row tile over B*T
    int row0 = rt << 4;

    const float* xrow = x + (size_t)(row0 + lanelo) * NC;
    const bf16*  wmat = wt + (size_t)m * (NH * NC);

    v8f acc[8];
#pragma unroll
    for (int j = 0; j < 8; ++j) acc[j] = (v8f){};

    for (int kk = 0; kk < NC / 32; ++kk) {
        int c0 = kk * 32;
        v16bf a = ldfragA_f32(xrow + c0 + hi * 8, xrow + c0 + 16 + hi * 8);
#pragma unroll
        for (int j = 0; j < 8; ++j) {
            const bf16* wrow = wmat + (size_t)(j * 16 + lanelo) * NC;
            v16bf b = ldfragB(wrow + c0, hi);
            acc[j] = wmma_bf16(a, b, acc[j]);
        }
    }

    if (m < 2) {
        bf16* outp = (m == 0) ? qb : kb;
#pragma unroll
        for (int j = 0; j < 8; ++j)
#pragma unroll
            for (int r = 0; r < 8; ++r)
                outp[(size_t)(row0 + r + 8 * hi) * NH + j * 16 + lanelo] =
                    (bf16)acc[j][r];
    } else {
        int b  = row0 >> 11;      // / T
        int t0 = row0 & 2047;
#pragma unroll
        for (int j = 0; j < 8; ++j)
#pragma unroll
            for (int r = 0; r < 8; ++r)
                vt[(size_t)(b * NH + j * 16 + lanelo) * NT + t0 + r + 8 * hi] =
                    (bf16)acc[j][r];
    }
}

// ---------------------------------------------------------------------------
// Kernel 3: per-wave flash attention. One wave = 16 query rows of one batch.
// 8*128 = 1024 waves -> 128 blocks of 256 threads. No workgroup barriers
// (waves have divergent causal trip counts); LDS slice is per-wave.
// ---------------------------------------------------------------------------
__global__ __launch_bounds__(256) void k_attn(const bf16* __restrict__ qb,
                                              const bf16* __restrict__ kb,
                                              const bf16* __restrict__ vt,
                                              float* __restrict__ out) {
    __shared__ __align__(16) bf16 pscr[8 * 16 * 40];  // 8 waves x (16 rows x 40)

    int widx   = threadIdx.x >> 5;
    int lane   = threadIdx.x & 31;
    int lanelo = lane & 15;
    int hi     = lane >> 4;

    int g  = blockIdx.x * 8 + widx;
    int b  = g >> 7;              // / (T/16)
    int q0 = (g & 127) << 4;

    const bf16* qrow  = qb + (size_t)(b * NT + q0 + lanelo) * NH;
    const bf16* kbase = kb + (size_t)b * NT * NH;
    const bf16* vbase = vt + (size_t)b * NH * NT;
    bf16*       pb    = pscr + widx * (16 * 40);

    // Q fragments (A operand): 16x128 over four K=32 chunks.
    v16bf Q[4];
#pragma unroll
    for (int f = 0; f < 4; ++f)
        Q[f] = ldfragA(qrow + f * 32 + hi * 8, qrow + f * 32 + 16 + hi * 8);

    float mrow[8], lrow[8];
    v8f   O[8];
#pragma unroll
    for (int r = 0; r < 8; ++r) { mrow[r] = -1e30f; lrow[r] = 0.0f; }
#pragma unroll
    for (int j = 0; j < 8; ++j) O[j] = (v8f){};

    const float scale = 0.08838834764831845f;  // 1/sqrt(128)
    int n_kt = (q0 + 47) >> 5;                 // causal: key tiles of 32

    for (int st = 0; st < n_kt; ++st) {
        int s0 = st * 32;
        // B operand for Q*K^T: column N = key row s0+lanelo (tile 0) /
        // s0+16+lanelo (tile 1); K = head dims f*32 + hi*8 .. +7, f*32 + 16 + hi*8 .. +7.
        const bf16* k0 = kbase + (size_t)(s0 + lanelo) * NH;
        const bf16* k1 = k0 + (size_t)16 * NH;
        if (st + 1 < n_kt)  // hint next K tile toward the caches
            __builtin_prefetch(kbase + (size_t)(s0 + 32 + lanelo) * NH, 0, 1);

        v8f S0 = (v8f){}, S1 = (v8f){};
#pragma unroll
        for (int f = 0; f < 4; ++f) {
            v16bf B0 = ldfragB(k0 + f * 32, hi);
            S0 = wmma_bf16(Q[f], B0, S0);
            v16bf B1 = ldfragB(k1 + f * 32, hi);
            S1 = wmma_bf16(Q[f], B1, S1);
        }

        // Online softmax. C layout: VGPR r, lane -> (row r+8*hi, col lanelo).
        float alpha[8];
#pragma unroll
        for (int r = 0; r < 8; ++r) {
            int qg = q0 + r + 8 * hi;
            int sg = s0 + lanelo;
            float a0 = S0[r] * scale; if (sg > qg)      a0 = -1e9f;
            float a1 = S1[r] * scale; if (sg + 16 > qg) a1 = -1e9f;
            float mn = fmaxf(mrow[r], rmax16(fmaxf(a0, a1)));
            alpha[r] = __expf(mrow[r] - mn);
            float p0 = __expf(a0 - mn);
            float p1 = __expf(a1 - mn);
            lrow[r] = lrow[r] * alpha[r] + rsum16(p0 + p1);
            mrow[r] = mn;
            int row = r + 8 * hi;                 // C layout -> A layout via LDS
            pb[row * 40 + lanelo]      = (bf16)p0;
            pb[row * 40 + 16 + lanelo] = (bf16)p1;
        }
#pragma unroll
        for (int j = 0; j < 8; ++j)
#pragma unroll
            for (int r = 0; r < 8; ++r) O[j][r] *= alpha[r];

        asm volatile("s_wait_dscnt 0" ::: "memory");
        // P as A operand (interleaved K split) from the per-wave LDS slice.
        v16bf P = ldfragA(pb + lanelo * 40 + hi * 8,
                          pb + lanelo * 40 + 16 + hi * 8);

        // O += P(16x32) @ V(32x16) per 16-wide head tile.
        // B operand: column N = head dim j*16+lanelo; K = s values
        // s0 + hi*8 .. +7 and s0 + 16 + hi*8 .. +7 (interleaved), from V^T rows.
#pragma unroll
        for (int j = 0; j < 8; ++j) {
            const bf16* vrow = vbase + (size_t)(j * 16 + lanelo) * NT + s0;
            v16bf V = ldfragB(vrow, hi);
            O[j] = wmma_bf16(P, V, O[j]);
        }
    }

    float inv[8];
#pragma unroll
    for (int r = 0; r < 8; ++r) inv[r] = 1.0f / lrow[r];
    float* orow = out + (size_t)(b * NT + q0) * NH;
#pragma unroll
    for (int j = 0; j < 8; ++j)
#pragma unroll
        for (int r = 0; r < 8; ++r)
            orow[(size_t)(r + 8 * hi) * NH + j * 16 + lanelo] = O[j][r] * inv[r];
}

// ---------------------------------------------------------------------------
extern "C" void kernel_launch(void* const* d_in, const int* in_sizes, int n_in,
                              void* d_out, int out_size, void* d_ws, size_t ws_size,
                              hipStream_t stream) {
    const float* x  = (const float*)d_in[0];
    const float* Wq = (const float*)d_in[1];
    const float* Wk = (const float*)d_in[2];
    const float* Wv = (const float*)d_in[3];
    float* out = (float*)d_out;

    bf16* wt = (bf16*)d_ws;               // [3][128][1024]
    bf16* qb = wt + WT_ELEMS;             // [B*T][128]
    bf16* kb = qb + QK_ELEMS;             // [B*T][128]
    bf16* vt = kb + QK_ELEMS;             // [B][128][T]  (V^T)

    k_wtrans<<<(WT_ELEMS + 255) / 256, 256, 0, stream>>>(Wq, Wk, Wv, wt);
    k_proj<<<(3 * 1024) / 8, 256, 0, stream>>>(x, wt, qb, kb, vt);
    k_attn<<<(NB * (NT / 16)) / 8, 256, 0, stream>>>(qb, kb, vt, out);
}
